// GNN_MyGAT_38208029065544
// MI455X (gfx1250) — compile-verified, matrix operand order checked by reading, not run on hardware
//
#include <hip/hip_runtime.h>
#include <hip/hip_bf16.h>
#include <cmath>

#define N_NODES 50000
#define N_EDGES 800000
#define DIM     128
#define AEDGE   4
#define HEADS   2
#define NGRAPH  128
#define NEG_SLOPE 0.2f
#define ATT_STRIDE (2*DIM + AEDGE)   // 260

typedef float  v8f   __attribute__((ext_vector_type(8)));
typedef __bf16 v8bf  __attribute__((ext_vector_type(8)));
typedef __bf16 v16bf __attribute__((ext_vector_type(16)));

__device__ __forceinline__ unsigned short f2bf(float f) {
    unsigned int u = __float_as_uint(f);
    u += 0x7FFFu + ((u >> 16) & 1u);        // round-to-nearest-even
    return (unsigned short)(u >> 16);
}

__device__ __forceinline__ float lrelu(float x) {
    return x > 0.f ? x : NEG_SLOPE * x;
}

__device__ __forceinline__ void atomicMaxFloat(float* addr, float val) {
    unsigned int* ua = (unsigned int*)addr;
    unsigned int old = *ua;
    while (__uint_as_float(old) < val) {
        unsigned int assumed = old;
        old = atomicCAS(ua, assumed, __float_as_uint(val));
        if (old == assumed) break;
    }
}

// ---------------- generic fill ----------------
__global__ void k_fill(float* __restrict__ p, float v, int n) {
    int i = blockIdx.x * blockDim.x + threadIdx.x;
    if (i < n) p[i] = v;
}

// ---------------- f32 -> bf16 cast ----------------
__global__ void k_cast(const float* __restrict__ in, unsigned short* __restrict__ out, int n) {
    int i = blockIdx.x * blockDim.x + threadIdx.x;
    if (i < n) out[i] = f2bf(in[i]);
}

// ---------------- weight transpose+cast: [128,256] row-major -> [256,128] n-major bf16 ----
__global__ void k_prep_w(const float* __restrict__ w, unsigned short* __restrict__ wt) {
    int i = blockIdx.x * blockDim.x + threadIdx.x;   // over 256*128
    if (i >= 256 * DIM) return;
    int nn = i / DIM, kk = i % DIM;
    wt[i] = f2bf(w[kk * 256 + nn]);
}

// ---------------- WMMA GEMM: C[M,256] = A[M,128](bf16) * B[128,256](bf16, stored n-major) + bias
// one wave per 16x16 output tile, K=128 -> 4x v_wmma_f32_16x16x32_bf16
__global__ void __launch_bounds__(256) k_gemm_bf16(
    const unsigned short* __restrict__ Abf,   // [M,128] row-major
    const unsigned short* __restrict__ Bt,    // [256,128] (col-of-C major, K contiguous)
    const float* __restrict__ bias,           // [256]
    float* __restrict__ C, int M)             // [M,256]
{
    const int lane = threadIdx.x & 31;
    const int wid  = blockIdx.x * (blockDim.x >> 5) + (threadIdx.x >> 5);
    const int mtile = wid >> 4;               // 256/16 = 16 n-tiles
    const int ntile = wid & 15;
    if (mtile * 16 >= M) return;              // wave-uniform: EXEC stays all-ones for WMMA
    const int l16 = lane & 15;
    const int sel = (lane < 16) ? 0 : 8;      // 16-bit A/B layout: hi half-wave holds K+8 / K+24

    const v8bf* arow = (const v8bf*)(Abf + (size_t)(mtile * 16 + l16) * DIM + sel);
    const v8bf* bcol = (const v8bf*)(Bt  + (size_t)(ntile * 16 + l16) * DIM + sel);

    v8f c = {};
#pragma unroll
    for (int ks = 0; ks < 4; ++ks) {          // K steps of 32
        v8bf alo = arow[ks * 4];              // K = 32*ks + sel .. +7
        v8bf ahi = arow[ks * 4 + 2];          // K = 32*ks + sel+16 .. +23
        v8bf blo = bcol[ks * 4];
        v8bf bhi = bcol[ks * 4 + 2];
        v16bf a = __builtin_shufflevector(alo, ahi, 0,1,2,3,4,5,6,7,8,9,10,11,12,13,14,15);
        v16bf b = __builtin_shufflevector(blo, bhi, 0,1,2,3,4,5,6,7,8,9,10,11,12,13,14,15);
        c = __builtin_amdgcn_wmma_f32_16x16x32_bf16(false, a, false, b, (short)0, c, false, false);
    }

    const int col   = ntile * 16 + l16;       // C layout: lane -> N
    const float bn  = bias[col];
    const int mbase = mtile * 16 + ((lane < 16) ? 0 : 8);  // VGPR v -> M = v (+8 for hi lanes)
#pragma unroll
    for (int v = 0; v < 8; ++v)
        C[(size_t)(mbase + v) * 256 + col] = c[v] + bn;
}

// ---------------- per-(node,head) attention dots ai/aj ----------------
__global__ void k_node_att(const float* __restrict__ xh, const float* __restrict__ att,
                           float* __restrict__ ai, float* __restrict__ aj) {
    int i = blockIdx.x * blockDim.x + threadIdx.x;   // over N*H
    if (i >= N_NODES * HEADS) return;
    int node = i >> 1, h = i & 1;
    const float* row = xh + (size_t)node * 256 + h * DIM;
    const float* a1  = att + h * ATT_STRIDE;
    const float* a2  = a1 + DIM;
    float si = 0.f, sj = 0.f;
    for (int d = 0; d < DIM; ++d) { si += row[d] * a1[d]; sj += row[d] * a2[d]; }
    ai[i] = si; aj[i] = sj;
}

// ---------------- per-edge: edge linear, raw alpha (leaky relu), segment max ----------------
__global__ void k_edge(const float* __restrict__ ein, const int* __restrict__ eidx,
                       const float* __restrict__ linE, const float* __restrict__ att,
                       const float* __restrict__ ai, const float* __restrict__ aj,
                       float* __restrict__ eh, float* __restrict__ alpha,
                       float* __restrict__ amax) {
    int e = blockIdx.x * blockDim.x + threadIdx.x;
    if (e >= N_EDGES) return;
    int src = eidx[e], dst = eidx[N_EDGES + e];
    float ev[AEDGE];
#pragma unroll
    for (int c = 0; c < AEDGE; ++c) ev[c] = ein[(size_t)e * AEDGE + c];
#pragma unroll
    for (int h = 0; h < HEADS; ++h) {
        float ae = 0.f;
#pragma unroll
        for (int c = 0; c < AEDGE; ++c) {
            float v = 0.f;
#pragma unroll
            for (int a = 0; a < AEDGE; ++a) v += ev[a] * linE[a * (HEADS * AEDGE) + h * AEDGE + c];
            eh[(size_t)e * 8 + h * AEDGE + c] = v;
            ae += v * att[h * ATT_STRIDE + 2 * DIM + c];
        }
        float ar = lrelu(ai[src * 2 + h] + aj[dst * 2 + h] + ae);
        alpha[(size_t)e * 2 + h] = ar;
        atomicMaxFloat(&amax[dst * 2 + h], ar);
    }
}

// ---------------- exp(alpha - max), accumulate denom ----------------
__global__ void k_edge_exp(const int* __restrict__ eidx, float* __restrict__ alpha,
                           const float* __restrict__ amax, float* __restrict__ denom) {
    int i = blockIdx.x * blockDim.x + threadIdx.x;   // over E*H
    if (i >= N_EDGES * HEADS) return;
    int e = i >> 1, h = i & 1;
    int dst = eidx[N_EDGES + e];
    float ex = expf(alpha[i] - amax[dst * 2 + h]);
    alpha[i] = ex;
    atomicAdd(&denom[dst * 2 + h], ex);
}

// ---------------- normalize alpha ----------------
__global__ void k_edge_norm(const int* __restrict__ eidx, float* __restrict__ alpha,
                            const float* __restrict__ denom) {
    int i = blockIdx.x * blockDim.x + threadIdx.x;   // over E*H
    if (i >= N_EDGES * HEADS) return;
    int e = i >> 1, h = i & 1;
    int dst = eidx[N_EDGES + e];
    alpha[i] = alpha[i] / denom[dst * 2 + h];
}

// ---------------- scatter-add messages: acc[dst] += alpha * xh[src] (4 floats/thread) ----
__global__ void k_edge_msg(const int* __restrict__ eidx, const float* __restrict__ alpha,
                           const float* __restrict__ xh, float* __restrict__ acc) {
    int i = blockIdx.x * blockDim.x + threadIdx.x;   // over E*64
    if (i >= N_EDGES * 64) return;
    int e  = i >> 6;
    int r4 = (i & 63) << 2;                          // 0..252, element offset within [H*D]=256
    int h  = r4 >> 7;
    int src = eidx[e], dst = eidx[N_EDGES + e];
    float a = alpha[(size_t)e * 2 + h];
    float4 xs = *(const float4*)(xh + (size_t)src * 256 + r4);
    float* ap = acc + (size_t)dst * 256 + r4;
    atomicAdd(ap + 0, a * xs.x);
    atomicAdd(ap + 1, a * xs.y);
    atomicAdd(ap + 2, a * xs.z);
    atomicAdd(ap + 3, a * xs.w);
}

// ---------------- node output: (+x skip), mean over heads, relu ----------------
__global__ void k_node_out(const float* __restrict__ acc, const float* __restrict__ xh,
                           float* __restrict__ hout) {
    int i = blockIdx.x * blockDim.x + threadIdx.x;   // over N*128
    if (i >= N_NODES * DIM) return;
    int n = i >> 7, d = i & 127;
    size_t base = (size_t)n * 256 + d;
    float v0 = acc[base] + xh[base];
    float v1 = acc[base + DIM] + xh[base + DIM];
    hout[i] = fmaxf(0.f, 0.5f * (v0 + v1));
}

// ---------------- edge output: mean over heads, relu ----------------
__global__ void k_edge_out(const float* __restrict__ eh, float* __restrict__ eout) {
    int i = blockIdx.x * blockDim.x + threadIdx.x;   // over E*4
    if (i >= N_EDGES * AEDGE) return;
    int e = i >> 2, c = i & 3;
    eout[i] = fmaxf(0.f, 0.5f * (eh[(size_t)e * 8 + c] + eh[(size_t)e * 8 + 4 + c]));
}

// ---------------- global add pool ----------------
__global__ void k_pool(const float* __restrict__ h, const int* __restrict__ batch,
                       float* __restrict__ pool) {
    int i = blockIdx.x * blockDim.x + threadIdx.x;   // over N*128
    if (i >= N_NODES * DIM) return;
    int n = i >> 7, d = i & 127;
    atomicAdd(&pool[batch[n] * DIM + d], h[i]);
}

// ---------------- final linear: y[g] = pool[g,:] . fw + fb ----------------
__global__ void k_final(const float* __restrict__ pool, const float* __restrict__ fw,
                        const float* __restrict__ fb, float* __restrict__ y) {
    int g = threadIdx.x;
    if (g >= NGRAPH) return;
    float s = 0.f;
    for (int d = 0; d < DIM; ++d) s += pool[g * DIM + d] * fw[d];
    y[g] = s + fb[0];
}

static inline int nblk(long long n, int bs) { return (int)((n + bs - 1) / bs); }

extern "C" void kernel_launch(void* const* d_in, const int* in_sizes, int n_in,
                              void* d_out, int out_size, void* d_ws, size_t ws_size,
                              hipStream_t stream) {
    const float* x     = (const float*)d_in[0];
    const int*   eidx  = (const int*)d_in[1];
    const float* eattr = (const float*)d_in[2];
    const int*   batch = (const int*)d_in[3];
    const float* fw    = (const float*)d_in[16];
    const float* fb    = (const float*)d_in[17];

    char* ws = (char*)d_ws;
    size_t off = 0;
    auto alloc = [&](size_t bytes) -> void* {
        void* p = ws + off;
        off = (off + bytes + 255) & ~(size_t)255;
        return p;
    };

    float*          hbuf0 = (float*)alloc(sizeof(float) * (size_t)N_NODES * DIM);
    float*          hbuf1 = (float*)alloc(sizeof(float) * (size_t)N_NODES * DIM);
    float*          ebuf0 = (float*)alloc(sizeof(float) * (size_t)N_EDGES * AEDGE);
    float*          ebuf1 = (float*)alloc(sizeof(float) * (size_t)N_EDGES * AEDGE);
    unsigned short* xb    = (unsigned short*)alloc(sizeof(unsigned short) * (size_t)N_NODES * DIM);
    unsigned short* wt    = (unsigned short*)alloc(sizeof(unsigned short) * 256 * DIM);
    float*          xh    = (float*)alloc(sizeof(float) * (size_t)N_NODES * 256);
    float*          acc   = (float*)alloc(sizeof(float) * (size_t)N_NODES * 256);
    float*          ai    = (float*)alloc(sizeof(float) * (size_t)N_NODES * HEADS);
    float*          aj    = (float*)alloc(sizeof(float) * (size_t)N_NODES * HEADS);
    float*          eh    = (float*)alloc(sizeof(float) * (size_t)N_EDGES * 8);
    float*          alpha = (float*)alloc(sizeof(float) * (size_t)N_EDGES * HEADS);
    float*          amax  = (float*)alloc(sizeof(float) * (size_t)N_NODES * HEADS);
    float*          denom = (float*)alloc(sizeof(float) * (size_t)N_NODES * HEADS);
    float*          pool  = (float*)alloc(sizeof(float) * (size_t)NGRAPH * DIM);
    (void)ws_size; (void)in_sizes; (void)n_in; (void)out_size;

    const int BS = 256;
    for (int l = 0; l < 3; ++l) {
        const float* linN = (const float*)d_in[4 + 4 * l];
        const float* linE = (const float*)d_in[5 + 4 * l];
        const float* att  = (const float*)d_in[6 + 4 * l];
        const float* bias = (const float*)d_in[7 + 4 * l];

        const float* hin = (l == 0) ? x     : ((l == 1) ? hbuf0 : hbuf1);
        const float* ein = (l == 0) ? eattr : ((l == 1) ? ebuf0 : ebuf1);
        float* hout = (l & 1) ? hbuf1 : hbuf0;
        float* eout = (l & 1) ? ebuf1 : ebuf0;

        k_cast  <<<nblk((long long)N_NODES * DIM, BS), BS, 0, stream>>>(hin, xb, N_NODES * DIM);
        k_prep_w<<<nblk(256 * DIM, BS), BS, 0, stream>>>(linN, wt);
        k_fill  <<<nblk((long long)N_NODES * 256, BS), BS, 0, stream>>>(acc, 0.f, N_NODES * 256);
        k_fill  <<<nblk(N_NODES * HEADS, BS), BS, 0, stream>>>(amax, -INFINITY, N_NODES * HEADS);
        k_fill  <<<nblk(N_NODES * HEADS, BS), BS, 0, stream>>>(denom, 0.f, N_NODES * HEADS);

        // one wave per 16x16 tile: (N/16)*16 waves, 8 waves/block
        int totalWaves = (N_NODES / 16) * (256 / 16);
        k_gemm_bf16<<<(totalWaves + 7) / 8, BS, 0, stream>>>(xb, wt, bias, xh, N_NODES);

        k_node_att <<<nblk(N_NODES * HEADS, BS), BS, 0, stream>>>(xh, att, ai, aj);
        k_edge     <<<nblk(N_EDGES, BS), BS, 0, stream>>>(ein, eidx, linE, att, ai, aj, eh, alpha, amax);
        k_edge_exp <<<nblk((long long)N_EDGES * HEADS, BS), BS, 0, stream>>>(eidx, alpha, amax, denom);
        k_edge_norm<<<nblk((long long)N_EDGES * HEADS, BS), BS, 0, stream>>>(eidx, alpha, denom);
        k_edge_msg <<<nblk((long long)N_EDGES * 64, BS), BS, 0, stream>>>(eidx, alpha, xh, acc);
        k_node_out <<<nblk((long long)N_NODES * DIM, BS), BS, 0, stream>>>(acc, xh, hout);
        k_edge_out <<<nblk((long long)N_EDGES * AEDGE, BS), BS, 0, stream>>>(eh, eout);
    }

    // final h lives in hbuf0 after layer 2
    k_fill <<<nblk(NGRAPH * DIM, BS), BS, 0, stream>>>(pool, 0.f, NGRAPH * DIM);
    k_pool <<<nblk((long long)N_NODES * DIM, BS), BS, 0, stream>>>(hbuf0, batch, pool);
    k_final<<<1, 128, 0, stream>>>(pool, fw, fb, (float*)d_out);
}
